// GraniteSpeechConformerAttention_4174708212097
// MI455X (gfx1250) — matrix operand order repaired and checked by reading, not compiled, not run on hardware
//
#include <hip/hip_runtime.h>
#include <math.h>

// ---------------------------------------------------------------------------
// Types for CDNA5 WMMA (wave32)
// ---------------------------------------------------------------------------
typedef __bf16 bf16_t;
typedef __attribute__((ext_vector_type(8)))  __bf16 v8bf;
typedef __attribute__((ext_vector_type(16))) __bf16 v16bf;
typedef __attribute__((ext_vector_type(8)))  float  v8f;

#define HID    1024
#define NHEAD  16
#define DHEAD  64
#define CTX    200
#define CPAD   208      // CTX padded to 13*16 for q/k row tiles
#define VPAD   224      // CTX padded to 7*32 for vT columns (PV K-chunks)
#define QRW    224      // width of per-row-tile QR strip (needs >= 215)
#define BB     8
#define TT     4000
#define NBLK   (TT/CTX) // 20
#define MTOK   (BB*TT)  // 32000
#define BNH    (BB*NBLK*NHEAD) // 2560
#define RELBASE 313     // rel_emb row for QR column 0 of row-tile 0

__device__ __forceinline__ v8f zero8() {
  v8f z = {0.f,0.f,0.f,0.f,0.f,0.f,0.f,0.f};
  return z;
}
__device__ __forceinline__ v16bf cat8(v8bf lo, v8bf hi) {
  return __builtin_shufflevector(lo, hi, 0,1,2,3,4,5,6,7,8,9,10,11,12,13,14,15);
}
// A fragment (16x32 bf16, row-major source). lane: m = base_row + (lane&15).
// Contiguous layout: k = 16h+e (matches ldB).  `row` points at (m, k0).
__device__ __forceinline__ v16bf ldA(const bf16_t* row, int h) {
  v8bf lo = *(const v8bf*)(row + 16*h);
  v8bf hi = *(const v8bf*)(row + 16*h + 8);
  return cat8(lo, hi);
}
// B fragment (32x16 bf16, N-major source). lane: n = base + (lane&15); k = 16h+e.
// `p` points at (n_row, k0 + 16*h): 16 contiguous bf16.
__device__ __forceinline__ v16bf ldB(const bf16_t* p) {
  v8bf lo = *(const v8bf*)(p);
  v8bf hi = *(const v8bf*)(p + 8);
  return cat8(lo, hi);
}
__device__ __forceinline__ v8f wmma_bf(v16bf a, v16bf b, v8f c) {
  // (neg_a, A, neg_b, B, c_mod, C, reuse_a, reuse_b)
  return __builtin_amdgcn_wmma_f32_16x16x32_bf16(false, a, false, b, (short)0, c, false, false);
}

// Async global -> LDS copy, 16 bytes per lane (GLOBAL_LOAD_ASYNC_TO_LDS_B128,
// ASYNCcnt-tracked).  ldsOff = byte offset within workgroup LDS (low 32 bits of
// the generic shared pointer, per the LDS aperture truncation rule).
__device__ __forceinline__ void async_cp16(unsigned ldsOff, const void* gaddr) {
  asm volatile("global_load_async_to_lds_b128 %0, %1, off"
               :
               : "v"(ldsOff), "v"(gaddr)
               : "memory");
}
__device__ __forceinline__ void async_wait0() {
  asm volatile("s_wait_asynccnt 0x0" ::: "memory");
}

// ---------------------------------------------------------------------------
// Kernel: pack weights (transposed, N-major) + rel_emb to bf16
// ---------------------------------------------------------------------------
__global__ __launch_bounds__(256) void pack_kernel(
    const float* __restrict__ wq, const float* __restrict__ wkv,
    const float* __restrict__ wo, const float* __restrict__ rel,
    bf16_t* __restrict__ wqkvT, bf16_t* __restrict__ woT, bf16_t* __restrict__ relb)
{
  const long N1 = 3072L*1024, N2 = 1024L*1024, N3 = 1025L*64;
  const long total = N1 + N2 + N3;
  for (long i = (long)blockIdx.x*256 + threadIdx.x; i < total; i += (long)gridDim.x*256) {
    if (i < N1) {
      int n = (int)(i >> 10), k = (int)(i & 1023);
      float v = (n < 1024) ? wq[(long)k*1024 + n] : wkv[(long)k*2048 + (n - 1024)];
      wqkvT[i] = (bf16_t)v;
    } else if (i < N1 + N2) {
      long j = i - N1;
      int n = (int)(j >> 10), k = (int)(j & 1023);
      woT[j] = (bf16_t)wo[(long)k*1024 + n];
    } else {
      long j = i - N1 - N2;
      relb[j] = (bf16_t)rel[j];
    }
  }
}

// ---------------------------------------------------------------------------
// Kernel: zero padding rows/cols of blocked q/k/vT buffers
// ---------------------------------------------------------------------------
__global__ __launch_bounds__(256) void padzero_kernel(
    bf16_t* __restrict__ qb, bf16_t* __restrict__ kb, bf16_t* __restrict__ vbT)
{
  const long bnh = blockIdx.x;
  const bf16_t z = (bf16_t)0.f;
  for (int i = threadIdx.x; i < (CPAD-CTX)*DHEAD; i += 256) {
    qb[bnh*CPAD*DHEAD + CTX*DHEAD + i] = z;
    kb[bnh*CPAD*DHEAD + CTX*DHEAD + i] = z;
  }
  for (int i = threadIdx.x; i < DHEAD*(VPAD-CTX); i += 256) {
    int d = i / (VPAD-CTX), c = CTX + i % (VPAD-CTX);
    vbT[bnh*DHEAD*VPAD + (long)d*VPAD + c] = z;
  }
}

// ---------------------------------------------------------------------------
// Kernel: LayerNorm (eps=1e-5) -> bf16 x, one block per token row
// ---------------------------------------------------------------------------
__global__ __launch_bounds__(256) void ln_kernel(
    const float* __restrict__ hs, const float* __restrict__ g,
    const float* __restrict__ bt, bf16_t* __restrict__ xb)
{
  const int row = blockIdx.x, tid = threadIdx.x;
  const float4 v = ((const float4*)(hs + (long)row*HID))[tid];
  float s  = v.x + v.y + v.z + v.w;
  float ss = v.x*v.x + v.y*v.y + v.z*v.z + v.w*v.w;
  #pragma unroll
  for (int off = 16; off > 0; off >>= 1) {
    s  += __shfl_down(s,  off, 32);
    ss += __shfl_down(ss, off, 32);
  }
  __shared__ float ls[8], lss[8], stats[2];
  const int w = tid >> 5, lane = tid & 31;
  if (lane == 0) { ls[w] = s; lss[w] = ss; }
  __syncthreads();
  if (tid == 0) {
    float a = 0.f, b2 = 0.f;
    #pragma unroll
    for (int i = 0; i < 8; i++) { a += ls[i]; b2 += lss[i]; }
    float mu  = a * (1.f/HID);
    float var = b2 * (1.f/HID) - mu*mu;
    stats[0] = mu; stats[1] = rsqrtf(var + 1e-5f);
  }
  __syncthreads();
  const float mu = stats[0], rs = stats[1];
  const float4 gg = ((const float4*)g)[tid];
  const float4 bb = ((const float4*)bt)[tid];
  bf16_t* o = xb + (long)row*HID + tid*4;
  o[0] = (bf16_t)((v.x - mu)*rs*gg.x + bb.x);
  o[1] = (bf16_t)((v.y - mu)*rs*gg.y + bb.y);
  o[2] = (bf16_t)((v.z - mu)*rs*gg.z + bb.z);
  o[3] = (bf16_t)((v.w - mu)*rs*gg.w + bb.w);
}

// ---------------------------------------------------------------------------
// Kernel: fused QKV GEMM (M=32000, N=3072, K=1024) -> blocked q/k, transposed v
//   grid (48 n-tiles, 250 m-tiles), block 256 (8 waves), wave = 16x64 tile
// ---------------------------------------------------------------------------
__global__ __launch_bounds__(256) void gemm_qkv_kernel(
    const bf16_t* __restrict__ xb, const bf16_t* __restrict__ wT,
    bf16_t* __restrict__ qb, bf16_t* __restrict__ kb, bf16_t* __restrict__ vbT)
{
  const int w = threadIdx.x >> 5, lane = threadIdx.x & 31;
  const int l15 = lane & 15, h = lane >> 4;
  const int m0 = blockIdx.y*128 + w*16;
  const int n0 = blockIdx.x*64;
  v8f acc[4] = {zero8(), zero8(), zero8(), zero8()};
  const bf16_t* arow = xb + (long)(m0 + l15)*HID;
  for (int k0 = 0; k0 < HID; k0 += 32) {
    v16bf a = ldA(arow + k0, h);
    #pragma unroll
    for (int t = 0; t < 4; t++) {
      v16bf b = ldB(wT + (long)(n0 + t*16 + l15)*HID + k0 + 16*h);
      acc[t] = wmma_bf(a, b, acc[t]);
    }
  }
  #pragma unroll
  for (int t = 0; t < 4; t++) {
    const int n = n0 + t*16 + l15;
    const int which = n >> 10;          // 0=q 1=k 2=v
    const int nn = n & 1023;
    const int hd = nn >> 6, d = nn & 63;
    #pragma unroll
    for (int r = 0; r < 8; r++) {
      const int m = m0 + r + 8*h;       // C-layout: M = reg + 8h
      const int bidx = m / TT, t2 = m % TT;
      const int blk = t2 / CTX, c = t2 % CTX;
      const long bnh = (long)(bidx*NBLK + blk)*NHEAD + hd;
      const bf16_t val = (bf16_t)acc[t][r];
      if (which == 0)      qb [bnh*CPAD*DHEAD + (long)c*DHEAD + d] = val;
      else if (which == 1) kb [bnh*CPAD*DHEAD + (long)c*DHEAD + d] = val;
      else                 vbT[bnh*DHEAD*VPAD + (long)d*VPAD  + c] = val;
    }
  }
}

// ---------------------------------------------------------------------------
// Kernel: blocked attention with Shaw relative-position bias.
//   One workgroup per (b, blk, head); wave handles 16-row tiles (13 tiles).
//   K and V^T staged once per workgroup into LDS with async-to-LDS copies
//   (ASYNCcnt), then QK^T and PV B-fragments come from ds_load_b128 (rel_emb ones from global).
//   pos[i,j] = q[i] . rel_emb[i-j+512]  (|i-j|<=199, clip never fires)
//   => QR = Q_tile @ relT (WMMA), staged in a per-wave LDS strip, gathered.
// ---------------------------------------------------------------------------
__global__ __launch_bounds__(256) void attn_kernel(
    const bf16_t* __restrict__ qb, const bf16_t* __restrict__ kb,
    const bf16_t* __restrict__ vbT, const bf16_t* __restrict__ relb,
    bf16_t* __restrict__ ao)
{
  __shared__ bf16_t Ks[CPAD*DHEAD];     // 26.0 KB  (K block, row-major [c][d])
  __shared__ bf16_t Vs[DHEAD*VPAD];     // 28.0 KB  (V^T block, [d][c])
  __shared__ bf16_t lds[8][16*QRW];     // 56.0 KB  per-wave QR strip / P tile
  const int w = threadIdx.x >> 5, lane = threadIdx.x & 31;
  const int l15 = lane & 15, h = lane >> 4;
  const int bnh = blockIdx.x;
  const int b = bnh / (NBLK*NHEAD);
  const int rem = bnh % (NBLK*NHEAD);
  const int blk = rem / NHEAD, head = rem % NHEAD;
  const bf16_t* Q  = qb  + (long)bnh*CPAD*DHEAD;
  const bf16_t* Kg = kb  + (long)bnh*CPAD*DHEAD;
  const bf16_t* VTg = vbT + (long)bnh*DHEAD*VPAD;
  bf16_t* buf = &lds[w][0];

  // ---- cooperative async stage of K and V^T into LDS ----
  {
    const unsigned ksOff = (unsigned)(uintptr_t)(&Ks[0]);
    const unsigned vsOff = (unsigned)(uintptr_t)(&Vs[0]);
    const int tid = threadIdx.x;
    const int KCH = CPAD*DHEAD/8;       // 16B chunks
    const int VCH = DHEAD*VPAD/8;
    for (int i = tid; i < KCH; i += 256)
      async_cp16(ksOff + (unsigned)i*16u, Kg + (long)i*8);
    for (int i = tid; i < VCH; i += 256)
      async_cp16(vsOff + (unsigned)i*16u, VTg + (long)i*8);
    async_wait0();
    __syncthreads();
  }

  for (int rt = w; rt < 13; rt += 8) {
    const int r0 = rt*16;
    // Q A-fragments for this row tile stay in registers the whole time.
    const bf16_t* qrow = Q + (long)(r0 + l15)*DHEAD;
    const v16bf aq0 = ldA(qrow, h);
    const v16bf aq1 = ldA(qrow + 32, h);

    // ---- QR strip: q[i] . rel_emb[RELBASE + r0 + c'], c' in [0,224) ----
    #pragma unroll 2
    for (int ct = 0; ct < 14; ct++) {
      const bf16_t* rrow = relb + (long)(RELBASE + r0 + ct*16 + l15)*DHEAD + 16*h;
      v8f acc = zero8();
      acc = wmma_bf(aq0, ldB(rrow),      acc);
      acc = wmma_bf(aq1, ldB(rrow + 32), acc);
      #pragma unroll
      for (int r = 0; r < 8; r++)
        buf[(r + 8*h)*QRW + ct*16 + l15] = (bf16_t)acc[r];
    }

    // ---- scores S = scale * (Q K^T + gather(QR)) ----
    v8f S[13];
    #pragma unroll
    for (int jt = 0; jt < 13; jt++) {
      const bf16_t* krow = Ks + (long)(jt*16 + l15)*DHEAD + 16*h;
      v8f acc = zero8();
      acc = wmma_bf(aq0, ldB(krow),      acc);
      acc = wmma_bf(aq1, ldB(krow + 32), acc);
      const int j = jt*16 + l15;
      #pragma unroll
      for (int r = 0; r < 8; r++) {
        const int il = r + 8*h;         // local row in tile
        float sv;
        if (j < CTX) {
          float pos = (float)buf[il*QRW + (il + 199 - j)];
          sv = (acc[r] + pos) * 0.125f; // DIM_HEAD^-0.5
        } else {
          sv = -INFINITY;               // padded key columns
        }
        acc[r] = sv;
      }
      S[jt] = acc;
    }

    // ---- softmax over j (row lives across 16 lanes of a half-wave) ----
    #pragma unroll
    for (int r = 0; r < 8; r++) {
      float mx = S[0][r];
      #pragma unroll
      for (int jt = 1; jt < 13; jt++) mx = fmaxf(mx, S[jt][r]);
      #pragma unroll
      for (int off = 1; off < 16; off <<= 1) mx = fmaxf(mx, __shfl_xor(mx, off, 32));
      float sum = 0.f;
      #pragma unroll
      for (int jt = 0; jt < 13; jt++) {
        float e = __expf(S[jt][r] - mx);
        S[jt][r] = e;
        sum += e;
      }
      #pragma unroll
      for (int off = 1; off < 16; off <<= 1) sum += __shfl_xor(sum, off, 32);
      const float inv = 1.f / sum;
      #pragma unroll
      for (int jt = 0; jt < 13; jt++) S[jt][r] *= inv;
    }

    // ---- P tile -> LDS (A-fragment layout source), pad cols 208..223 = 0 ----
    #pragma unroll
    for (int r = 0; r < 8; r++)
      buf[(r + 8*h)*QRW + 208 + l15] = (bf16_t)0.f;
    #pragma unroll
    for (int jt = 0; jt < 13; jt++) {
      #pragma unroll
      for (int r = 0; r < 8; r++)
        buf[(r + 8*h)*QRW + jt*16 + l15] = (bf16_t)S[jt][r];
    }

    // ---- O = P @ V  (contract j in chunks of 32, LDS V^T gives contiguous B) ----
    v8f O[4] = {zero8(), zero8(), zero8(), zero8()};
    #pragma unroll
    for (int ko = 0; ko < 7; ko++) {
      const bf16_t* prow = buf + (long)l15*QRW + ko*32;
      v16bf ap = ldA(prow, h);
      #pragma unroll
      for (int t = 0; t < 4; t++) {
        const bf16_t* vrow = Vs + (long)(t*16 + l15)*VPAD + ko*32 + 16*h;
        O[t] = wmma_bf(ap, ldB(vrow), O[t]);
      }
    }
    #pragma unroll
    for (int t = 0; t < 4; t++) {
      #pragma unroll
      for (int r = 0; r < 8; r++) {
        const int i = r0 + r + 8*h;
        if (i < CTX) {
          const long tok = (long)b*TT + (long)blk*CTX + i;
          ao[tok*HID + head*DHEAD + t*16 + l15] = (bf16_t)O[t][r];
        }
      }
    }
  }
}

// ---------------------------------------------------------------------------
// Kernel: output projection (M=32000, N=1024, K=1024), fp32 out + bias
// ---------------------------------------------------------------------------
__global__ __launch_bounds__(256) void gemm_out_kernel(
    const bf16_t* __restrict__ ao, const bf16_t* __restrict__ woT,
    const float* __restrict__ bo, float* __restrict__ out)
{
  const int w = threadIdx.x >> 5, lane = threadIdx.x & 31;
  const int l15 = lane & 15, h = lane >> 4;
  const int m0 = blockIdx.y*128 + w*16;
  const int n0 = blockIdx.x*64;
  v8f acc[4] = {zero8(), zero8(), zero8(), zero8()};
  const bf16_t* arow = ao + (long)(m0 + l15)*HID;
  for (int k0 = 0; k0 < HID; k0 += 32) {
    v16bf a = ldA(arow + k0, h);
    #pragma unroll
    for (int t = 0; t < 4; t++) {
      v16bf b = ldB(woT + (long)(n0 + t*16 + l15)*HID + k0 + 16*h);
      acc[t] = wmma_bf(a, b, acc[t]);
    }
  }
  #pragma unroll
  for (int t = 0; t < 4; t++) {
    const int n = n0 + t*16 + l15;
    const float bias = bo[n];
    #pragma unroll
    for (int r = 0; r < 8; r++) {
      const int m = m0 + r + 8*h;
      out[(long)m*HID + n] = acc[t][r] + bias;
    }
  }
}

// ---------------------------------------------------------------------------
// Host launcher
// ---------------------------------------------------------------------------
extern "C" void kernel_launch(void* const* d_in, const int* in_sizes, int n_in,
                              void* d_out, int out_size, void* d_ws, size_t ws_size,
                              hipStream_t stream)
{
  const float* hs   = (const float*)d_in[0];
  const float* ln_w = (const float*)d_in[1];
  const float* ln_b = (const float*)d_in[2];
  const float* wq   = (const float*)d_in[3];
  const float* wkv  = (const float*)d_in[4];
  const float* wo   = (const float*)d_in[5];
  const float* bo   = (const float*)d_in[6];
  const float* rel  = (const float*)d_in[7];

  char* p = (char*)d_ws;
  auto alloc = [&](size_t bytes) -> char* {
    char* r = p;
    p += (bytes + 255) & ~(size_t)255;
    return r;
  };
  bf16_t* xb    = (bf16_t*)alloc((size_t)MTOK*HID*2);          // normalized x
  bf16_t* wqkvT = (bf16_t*)alloc((size_t)3072*1024*2);         // fused W^T
  bf16_t* woT   = (bf16_t*)alloc((size_t)1024*1024*2);
  bf16_t* relb  = (bf16_t*)alloc((size_t)1025*64*2);
  bf16_t* qb    = (bf16_t*)alloc((size_t)BNH*CPAD*DHEAD*2);    // blocked q
  bf16_t* kb    = (bf16_t*)alloc((size_t)BNH*CPAD*DHEAD*2);    // blocked k
  bf16_t* vbT   = (bf16_t*)alloc((size_t)BNH*DHEAD*VPAD*2);    // blocked v^T
  bf16_t* ao    = (bf16_t*)alloc((size_t)MTOK*HID*2);          // attention out

  pack_kernel   <<<16641, 256, 0, stream>>>(wq, wkv, wo, rel, wqkvT, woT, relb);
  padzero_kernel<<<BNH,   256, 0, stream>>>(qb, kb, vbT);
  ln_kernel     <<<MTOK,  256, 0, stream>>>(hs, ln_w, ln_b, xb);
  gemm_qkv_kernel<<<dim3(48, 250), 256, 0, stream>>>(xb, wqkvT, qb, kb, vbT);
  attn_kernel   <<<BNH,   256, 0, stream>>>(qb, kb, vbT, relb, ao);
  gemm_out_kernel<<<dim3(16, 250), 256, 0, stream>>>(ao, woT, bo, (float*)d_out);
}
